// RegionProposalNetwork_40793599377901
// MI455X (gfx1250) — matrix operand order repaired and checked by reading, not compiled, not run on hardware
//
#include <hip/hip_runtime.h>
#include <hip/hip_bf16.h>
#include <math.h>

typedef _Float16 v8h  __attribute__((ext_vector_type(8)));
typedef _Float16 v16h __attribute__((ext_vector_type(16)));
typedef float    v8f  __attribute__((ext_vector_type(8)));

#define CIN   512
#define COUT  512
#define HH    62
#define WW    62
#define BB    4
#define HW    3844        // 62*62
#define NTOT  15376       // 4*3844
#define KTOT  4608        // 512*9
#define NANC  34596       // 3844*9
#define PRE_K 1000
#define NMS_TH 0.7f
#define CLAMPV 4.135166556742356f   // log(1000/16)
#define MIN_SIZE 0.001f

// ---- CDNA5 async global->LDS path (probe-guarded) ----
#if defined(__has_builtin)
#if __has_builtin(__builtin_amdgcn_global_load_async_to_lds_b128)
#define USE_ASYNC_LDS 1
#endif
#endif

#if defined(USE_ASYNC_LDS)
// exact param types per clang diagnostic: (addrspace(1) int4*, addrspace(3) int4*, imm offset, imm cpol)
typedef int ivec4 __attribute__((vector_size(4 * sizeof(int))));
typedef __attribute__((address_space(1))) ivec4 gvec4_t;
typedef __attribute__((address_space(3))) ivec4 lvec4_t;
#endif

#if defined(__has_builtin)
#if __has_builtin(__builtin_amdgcn_s_wait_asynccnt)
#define WAIT_ASYNC() __builtin_amdgcn_s_wait_asynccnt(0)
#else
#define WAIT_ASYNC() asm volatile("s_wait_asynccnt 0x0" ::: "memory")
#endif
#else
#define WAIT_ASYNC() asm volatile("s_wait_asynccnt 0x0" ::: "memory")
#endif

// ------------------------------------------------------------------
// f32 -> f16 conversion (one-time; halves HBM traffic for the GEMM)
// ------------------------------------------------------------------
__global__ void k_cvt_f16(const float* __restrict__ src, _Float16* __restrict__ dst, int n) {
  int i = blockIdx.x * blockDim.x + threadIdx.x;
  if (i < n) dst[i] = (_Float16)src[i];
}

// ------------------------------------------------------------------
// 3x3 conv as implicit GEMM:  C[M=512][N=15376] = W[M][K=4608] * X[K][N]
// Block tile 128x128x32, 8 wave32 waves, wave tile 64x32 (4x2 WMMA tiles).
// Double-buffered LDS, one barrier per K-step, next-tile staging issued
// between barrier and WMMAs so VMEM overlaps the matrix pipe. Weight tile
// staged via CDNA5 async global->LDS (ASYNCcnt-tracked, bypasses VGPRs).
// ------------------------------------------------------------------
#define BM 128
#define BN 128
#define BK 32
#define SA 40   // padded LDS row stride in halves (80B, 16B aligned)
#define KSTEPS (KTOT / BK)   // 144

__global__ __launch_bounds__(256)
void k_conv3x3_wmma(const _Float16* __restrict__ wf, const _Float16* __restrict__ xf,
                    const float* __restrict__ bias, _Float16* __restrict__ out) {
  __shared__ _Float16 As[2][BM * SA];   // weights tile  [m][k]
  __shared__ _Float16 Bs[2][BN * SA];   // patches tile  [n][k] (K-contiguous)

  const int tid  = threadIdx.x;
  const int wave = tid >> 5;
  const int lane = tid & 31;
  const int lm   = lane & 15;
  const int hi   = lane >> 4;
  const int wm   = wave >> 2;        // 0..1  (M direction)
  const int wn   = wave & 3;         // 0..3  (N direction)
  const int m0   = blockIdx.y * BM;
  const int n0   = blockIdx.x * BN;

  v8f acc[4][2];
  #pragma unroll
  for (int a = 0; a < 4; ++a)
    #pragma unroll
    for (int b = 0; b < 2; ++b)
      #pragma unroll
      for (int r = 0; r < 8; ++r) acc[a][b][r] = 0.0f;

  // ---- A staging role: 32B (16 halves) per thread, contiguous in K ----
  const int am = tid >> 1;           // weight row 0..127
  const int ac = tid & 1;            // 16-half chunk

  // ---- B staging role: ONE pixel per thread, 16 K-values per thread ----
  const int nB  = n0 + (tid & 127);
  const int okN = (nB < NTOT) ? 1 : 0;
  const int nCl = okN ? nB : (NTOT - 1);
  const int bI  = nCl / HW;
  const int pI  = nCl - bI * HW;
  const int yI  = pI / WW;
  const int xI  = pI - yI * WW;
  const _Float16* __restrict__ srcB = xf + (size_t)bI * CIN * HW;
  const int kc = (tid >> 7) << 4;    // 0 or 16: which half of the K-tile

  // rolling decomposition of k_start = k0 + kc  (advances by BK=32 each call:
  // r9 += 5 mod 9, ci += 3 (+carry) -- no divides inside the loop)
  int ci0 = kc / 9;
  int r90 = kc - ci0 * 9;

  auto stageA = [&](int k0, int bufi) {
    const _Float16* src = wf + (size_t)(m0 + am) * KTOT + k0 + ac * 16;
    _Float16* dst = &As[bufi][am * SA + ac * 16];
    if (k0 + BK < KTOT) __builtin_prefetch(src + BK, 0, 0);   // global_prefetch
#if defined(USE_ASYNC_LDS)
    __builtin_amdgcn_global_load_async_to_lds_b128(
        (gvec4_t*)src, (lvec4_t*)dst, 0, 0);
    __builtin_amdgcn_global_load_async_to_lds_b128(
        (gvec4_t*)(src + 8), (lvec4_t*)(dst + 8), 0, 0);
#else
    uint4 p0 = *(const uint4*)(src);
    uint4 p1 = *(const uint4*)(src + 8);
    uint4* d = (uint4*)dst;
    d[0] = p0; d[1] = p1;
#endif
  };

  auto stageB = [&](int bufi) {
    int ciL = ci0;
    int kyL = r90 / 3;
    int kxL = r90 - kyL * 3;
    alignas(16) _Float16 buf[16];
    #pragma unroll
    for (int kk = 0; kk < 16; ++kk) {
      const int iy = yI + kyL - 1;
      const int ix = xI + kxL - 1;
      const int ok = okN & ((unsigned)iy < (unsigned)HH) & ((unsigned)ix < (unsigned)WW);
      const int iyc = (iy < 0) ? 0 : (iy > HH - 1 ? HH - 1 : iy);
      const int ixc = (ix < 0) ? 0 : (ix > WW - 1 ? WW - 1 : ix);
      const _Float16 v = srcB[(size_t)ciL * HW + iyc * WW + ixc];  // always in-bounds
      buf[kk] = ok ? v : (_Float16)0.0f;                            // cndmask, no exec branch
      ++kxL;
      if (kxL == 3) { kxL = 0; ++kyL; if (kyL == 3) { kyL = 0; ++ciL; } }
    }
    _Float16* row = &Bs[bufi][(tid & 127) * SA + kc];
    *(v8h*)(row)     = *(const v8h*)&buf[0];
    *(v8h*)(row + 8) = *(const v8h*)&buf[8];
    // advance rolling k-decomposition by BK=32
    r90 += 5; ci0 += 3;
    if (r90 >= 9) { r90 -= 9; ++ci0; }
  };

  stageA(0, 0);
  stageB(0);

  for (int kt = 0; kt < KSTEPS; ++kt) {
    const int cur = kt & 1;
#if defined(USE_ASYNC_LDS)
    WAIT_ASYNC();
#endif
    __syncthreads();                       // staged tile `cur` is ready

    if (kt + 1 < KSTEPS) {                 // issue next tile early: overlaps WMMAs
      stageA((kt + 1) * BK, 1 - cur);
      stageB(1 - cur);
    }

    // ---- fragments per ISA 16-bit layouts, then 8 WMMAs ----
    v16h af[4], bf[2];
    #pragma unroll
    for (int mt = 0; mt < 4; ++mt) {
      const int mr = wm * 64 + mt * 16 + lm;
      const v8h lo = *(const v8h*)&As[cur][mr * SA + hi * 8];       // K = kb..kb+7
      const v8h hh = *(const v8h*)&As[cur][mr * SA + hi * 8 + 16];  // K = kb+16..kb+23
      af[mt] = __builtin_shufflevector(lo, hh, 0,1,2,3,4,5,6,7,8,9,10,11,12,13,14,15);
    }
    #pragma unroll
    for (int nt = 0; nt < 2; ++nt) {
      const int nr = wn * 32 + nt * 16 + lm;
      const v8h lo = *(const v8h*)&Bs[cur][nr * SA + hi * 8];       // K = kb2..kb2+7
      const v8h hh = *(const v8h*)&Bs[cur][nr * SA + hi * 8 + 16];  // K = kb2+16..kb2+23
      bf[nt] = __builtin_shufflevector(lo, hh, 0,1,2,3,4,5,6,7,8,9,10,11,12,13,14,15);
    }
    #pragma unroll
    for (int mt = 0; mt < 4; ++mt)
      #pragma unroll
      for (int nt = 0; nt < 2; ++nt)
        acc[mt][nt] = __builtin_amdgcn_wmma_f32_16x16x32_f16(
            false, af[mt], false, bf[nt], (short)0, acc[mt][nt], false, false);
  }

  // ---- epilogue: bias + ReLU, store f16 (feeds the 1x1 convs) ----
  #pragma unroll
  for (int mt = 0; mt < 4; ++mt) {
    #pragma unroll
    for (int nt = 0; nt < 2; ++nt) {
      const int n = n0 + wn * 32 + nt * 16 + lm;
      if (n < NTOT) {
        #pragma unroll
        for (int r = 0; r < 8; ++r) {
          const int m = m0 + wm * 64 + mt * 16 + r + hi * 8;
          const float v = fmaxf(acc[mt][nt][r] + bias[m], 0.0f);
          out[(size_t)m * NTOT + n] = (_Float16)v;
        }
      }
    }
  }
}

// ------------------------------------------------------------------
// Fused 1x1 convs: 45 output channels (9 conf + 36 bbox), weights in LDS
// ------------------------------------------------------------------
__global__ __launch_bounds__(256)
void k_conv1x1(const _Float16* __restrict__ t, const float* __restrict__ confw,
               const float* __restrict__ confb, const float* __restrict__ bbw,
               const float* __restrict__ bbb, float* __restrict__ cb) {
  __shared__ _Float16 wsm[45 * 512];   // 46 KB
  for (int i = threadIdx.x; i < 45 * 512; i += blockDim.x) {
    const int j = i >> 9, c = i & 511;
    const float w = (j < 9) ? confw[j * 512 + c] : bbw[(j - 9) * 512 + c];
    wsm[i] = (_Float16)w;
  }
  __syncthreads();
  const int n = blockIdx.x * blockDim.x + threadIdx.x;
  if (n >= NTOT) return;
  float acc[45];
  #pragma unroll
  for (int j = 0; j < 45; ++j) acc[j] = 0.0f;
  for (int c = 0; c < 512; ++c) {
    const float v = (float)t[(size_t)c * NTOT + n];
    #pragma unroll
    for (int j = 0; j < 45; ++j) acc[j] += v * (float)wsm[j * 512 + c];
  }
  #pragma unroll
  for (int j = 0; j < 45; ++j) {
    const float bs = (j < 9) ? confb[j] : bbb[j - 9];
    cb[(size_t)j * NTOT + n] = acc[j] + bs;
  }
}

// ------------------------------------------------------------------
// Anchor decode + sigmoid scores
// ------------------------------------------------------------------
__global__ void k_decode(const float* __restrict__ cb, float* __restrict__ scores,
                         float* __restrict__ prop) {
  const int g = blockIdx.x * blockDim.x + threadIdx.x;
  if (g >= BB * NANC) return;
  const int b = g / NANC;
  const int i = g - b * NANC;
  const int p = i / 9;
  const int a = i - p * 9;
  const int n = b * HW + p;
  const int si = a / 3;
  const int ri = a - si * 3;
  const float size = (si == 0) ? 128.0f : (si == 1) ? 256.0f : 512.0f;
  const float sr   = (ri == 0) ? 0.70710678118654752f : (ri == 1) ? 1.0f : 1.41421356237309505f;
  const float aw = size / sr;
  const float ah = size * sr;
  const int x = p % WW, y = p / WW;
  const float cx = (x + 0.5f) * 16.0f;
  const float cy = (y + 0.5f) * 16.0f;
  const float conf = cb[(size_t)a * NTOT + n];
  const float o0 = cb[(size_t)(9 + a * 4 + 0) * NTOT + n];
  const float o1 = cb[(size_t)(9 + a * 4 + 1) * NTOT + n];
  const float o2 = cb[(size_t)(9 + a * 4 + 2) * NTOT + n];
  const float o3 = cb[(size_t)(9 + a * 4 + 3) * NTOT + n];
  const float xc = cx + o0 * aw;
  const float yc = cy + o1 * ah;
  const float pw = expf(fminf(o2, CLAMPV)) * aw;
  const float ph = expf(fminf(o3, CLAMPV)) * ah;
  scores[g] = 1.0f / (1.0f + expf(-conf));
  float* pr = prop + (size_t)g * 4;
  pr[0] = xc - 0.5f * pw;
  pr[1] = yc - 0.5f * ph;
  pr[2] = xc + 0.5f * pw;
  pr[3] = yc + 0.5f * ph;
}

// ------------------------------------------------------------------
// Deterministic per-batch top-1000: histogram select + stable compaction
// + full bitonic sort of <=4096 candidates (exact score order).
// ------------------------------------------------------------------
__device__ __forceinline__ unsigned mkey(float s) {
  const unsigned u = __float_as_uint(s);
  return (u & 0x80000000u) ? ~u : (u | 0x80000000u);
}

__global__ __launch_bounds__(1024)
void k_topk(const float* __restrict__ scores, const float* __restrict__ prop,
            const int* __restrict__ imh, const int* __restrict__ imw,
            float* __restrict__ topS, float* __restrict__ topBox,
            int* __restrict__ topValid) {
  __shared__ int hist[4096];
  __shared__ unsigned long long cand[4096];
  __shared__ int sc[1024];
  __shared__ int sh_bin, sh_base;
  const int b = blockIdx.x;
  const int tid = threadIdx.x;
  const float* sb = scores + (size_t)b * NANC;

  for (int i = tid; i < 4096; i += 1024) { hist[i] = 0; cand[i] = 0ull; }
  __syncthreads();
  for (int i = tid; i < NANC; i += 1024)
    atomicAdd(&hist[mkey(sb[i]) >> 20], 1);
  __syncthreads();
  if (tid == 0) {
    int acc = 0, bin = 4095;
    for (; bin > 0; --bin) { acc += hist[bin]; if (acc >= PRE_K) break; }
    sh_bin = bin; sh_base = 0;
  }
  __syncthreads();
  const unsigned thr = (unsigned)sh_bin;

  for (int start = 0; start < NANC; start += 1024) {
    const int i = start + tid;
    unsigned key = 0; int flag = 0;
    if (i < NANC) { key = mkey(sb[i]); flag = ((key >> 20) >= thr) ? 1 : 0; }
    sc[tid] = flag; __syncthreads();
    for (int off = 1; off < 1024; off <<= 1) {
      const int v = (tid >= off) ? sc[tid - off] : 0;
      __syncthreads();
      sc[tid] += v;
      __syncthreads();
    }
    const int pos = sh_base + sc[tid] - flag;
    if (flag && pos < 4096)
      cand[pos] = ((unsigned long long)key << 32) |
                  (unsigned long long)(0xFFFFFFFFu - (unsigned)i);
    __syncthreads();
    if (tid == 1023) sh_base += sc[1023];
    __syncthreads();
  }

  // bitonic descending sort, ties -> smallest index first
  for (int k2 = 2; k2 <= 4096; k2 <<= 1) {
    for (int j = k2 >> 1; j > 0; j >>= 1) {
      __syncthreads();
      for (int t = tid; t < 4096; t += 1024) {
        const int ixj = t ^ j;
        if (ixj > t) {
          const unsigned long long va = cand[t], vb = cand[ixj];
          const bool descBlk = ((t & k2) == 0);
          if (descBlk ? (va < vb) : (va > vb)) { cand[t] = vb; cand[ixj] = va; }
        }
      }
    }
  }
  __syncthreads();

  const float fw = (float)imw[0];
  const float fh = (float)imh[0];
  if (tid < PRE_K) {
    const unsigned long long v = cand[tid];
    const unsigned idx = 0xFFFFFFFFu - (unsigned)(v & 0xFFFFFFFFull);
    const float s = sb[idx];
    const float* pr = prop + ((size_t)b * NANC + idx) * 4;
    const float x1 = fminf(fmaxf(pr[0], 0.0f), fw);
    const float y1 = fminf(fmaxf(pr[1], 0.0f), fh);
    const float x2 = fminf(fmaxf(pr[2], 0.0f), fw);
    const float y2 = fminf(fmaxf(pr[3], 0.0f), fh);
    const int valid = (x2 - x1 >= MIN_SIZE) && (y2 - y1 >= MIN_SIZE) && (s >= 0.0f);
    const int o = b * PRE_K + tid;
    topS[o] = s;
    topValid[o] = valid;
    topBox[o * 4 + 0] = x1; topBox[o * 4 + 1] = y1;
    topBox[o * 4 + 2] = x2; topBox[o * 4 + 3] = y2;
  }
}

// ------------------------------------------------------------------
// Sequential NMS (one block per batch) + stable compaction into d_out
// ------------------------------------------------------------------
__global__ __launch_bounds__(1024)
void k_nms(const float* __restrict__ topS, const float* __restrict__ topBox,
           const int* __restrict__ topValid, float* __restrict__ outAll) {
  __shared__ float bx[PRE_K * 4];
  __shared__ int keep[1024];
  __shared__ int sc[1024];
  const int b = blockIdx.x;
  const int tid = threadIdx.x;

  if (tid < PRE_K) {
    const int o = b * PRE_K + tid;
    bx[tid * 4 + 0] = topBox[o * 4 + 0];
    bx[tid * 4 + 1] = topBox[o * 4 + 1];
    bx[tid * 4 + 2] = topBox[o * 4 + 2];
    bx[tid * 4 + 3] = topBox[o * 4 + 3];
    keep[tid] = topValid[o];
  } else {
    keep[tid] = 0;
  }
  __syncthreads();

  float mx1 = 0, my1 = 0, mx2 = 0, my2 = 0, mya = 0;
  if (tid < PRE_K) {
    mx1 = bx[tid * 4 + 0]; my1 = bx[tid * 4 + 1];
    mx2 = bx[tid * 4 + 2]; my2 = bx[tid * 4 + 3];
    mya = (mx2 - mx1) * (my2 - my1);
  }
  for (int i = 0; i < PRE_K; ++i) {
    __syncthreads();
    if (tid < PRE_K && tid > i && keep[i] && keep[tid]) {
      const float ix1 = bx[i * 4 + 0], iy1 = bx[i * 4 + 1];
      const float ix2 = bx[i * 4 + 2], iy2 = bx[i * 4 + 3];
      const float lx = fmaxf(ix1, mx1), ly = fmaxf(iy1, my1);
      const float rx = fminf(ix2, mx2), ry = fminf(iy2, my2);
      const float iw = fmaxf(rx - lx, 0.0f), ih = fmaxf(ry - ly, 0.0f);
      const float inter = iw * ih;
      const float uni = (ix2 - ix1) * (iy2 - iy1) + mya - inter;
      if (inter / fmaxf(uni, 1e-9f) > NMS_TH) keep[tid] = 0;
    }
  }
  __syncthreads();

  const int flag = (tid < PRE_K) ? keep[tid] : 0;
  sc[tid] = flag; __syncthreads();
  for (int off = 1; off < 1024; off <<= 1) {
    const int v = (tid >= off) ? sc[tid - off] : 0;
    __syncthreads();
    sc[tid] += v;
    __syncthreads();
  }
  const int pos = sc[tid] - flag;

  float* fboxes  = outAll;                     // [4][1000][4]
  float* fscores = outAll + BB * PRE_K * 4;    // [4][1000]
  for (int q = tid; q < PRE_K * 4; q += 1024) fboxes[b * PRE_K * 4 + q] = 0.0f;
  if (tid < PRE_K) fscores[b * PRE_K + tid] = 0.0f;
  __syncthreads();
  if (flag) {
    fboxes[((size_t)b * PRE_K + pos) * 4 + 0] = mx1;
    fboxes[((size_t)b * PRE_K + pos) * 4 + 1] = my1;
    fboxes[((size_t)b * PRE_K + pos) * 4 + 2] = mx2;
    fboxes[((size_t)b * PRE_K + pos) * 4 + 3] = my2;
    fscores[(size_t)b * PRE_K + pos] = topS[b * PRE_K + tid];
  }
}

// ------------------------------------------------------------------
extern "C" void kernel_launch(void* const* d_in, const int* in_sizes, int n_in,
                              void* d_out, int out_size, void* d_ws, size_t ws_size,
                              hipStream_t stream) {
  const float* x   = (const float*)d_in[0];
  const float* w1  = (const float*)d_in[1];
  const float* b1  = (const float*)d_in[2];
  const float* cw  = (const float*)d_in[3];
  const float* cbi = (const float*)d_in[4];
  const float* bw  = (const float*)d_in[5];
  const float* bbi = (const float*)d_in[6];
  const int*   imh = (const int*)d_in[7];
  const int*   imw = (const int*)d_in[8];

  char* ws = (char*)d_ws;
  size_t off = 0;
  auto alloc = [&](size_t bytes) -> void* {
    void* p = ws + off;
    off += (bytes + 255) & ~(size_t)255;
    return p;
  };
  _Float16* wf16   = (_Float16*)alloc((size_t)COUT * KTOT * sizeof(_Float16));
  _Float16* inf16  = (_Float16*)alloc((size_t)BB * CIN * HW * sizeof(_Float16));
  _Float16* tf16   = (_Float16*)alloc((size_t)COUT * NTOT * sizeof(_Float16));
  float*    cb     = (float*)alloc((size_t)45 * NTOT * sizeof(float));
  float*    scores = (float*)alloc((size_t)BB * NANC * sizeof(float));
  float*    prop   = (float*)alloc((size_t)BB * NANC * 4 * sizeof(float));
  float*    topS   = (float*)alloc((size_t)BB * PRE_K * sizeof(float));
  float*    topBox = (float*)alloc((size_t)BB * PRE_K * 4 * sizeof(float));
  int*      topVal = (int*)alloc((size_t)BB * PRE_K * sizeof(int));

  const int nw = COUT * KTOT;
  k_cvt_f16<<<(nw + 255) / 256, 256, 0, stream>>>(w1, wf16, nw);
  const int nx = BB * CIN * HW;
  k_cvt_f16<<<(nx + 255) / 256, 256, 0, stream>>>(x, inf16, nx);

  dim3 gconv((NTOT + BN - 1) / BN, COUT / BM);   // (121, 4)
  k_conv3x3_wmma<<<gconv, 256, 0, stream>>>(wf16, inf16, b1, tf16);

  k_conv1x1<<<(NTOT + 255) / 256, 256, 0, stream>>>(tf16, cw, cbi, bw, bbi, cb);

  const int nd = BB * NANC;
  k_decode<<<(nd + 255) / 256, 256, 0, stream>>>(cb, scores, prop);

  k_topk<<<BB, 1024, 0, stream>>>(scores, prop, imh, imw, topS, topBox, topVal);
  k_nms<<<BB, 1024, 0, stream>>>(topS, topBox, topVal, (float*)d_out);
}
